// BidGRU_52080773431933
// MI455X (gfx1250) — compile-verified, matrix operand order checked by reading, not run on hardware
//
#include <hip/hip_runtime.h>
#include <hip/hip_bf16.h>

// ---------------------------------------------------------------------------
// Bidirectional GRU for MI455X (gfx1250), wave32 + v_wmma_f32_16x16x32_bf16,
// TDM (tensor_load_to_lds) double-buffered staging in the projection GEMM.
// I=512, H=1024, B=64, T=512.
// ---------------------------------------------------------------------------

#define GI 512
#define GH 1024
#define GB 64
#define GT 512
#define TBH ((size_t)GT * GB * GH)   // 33,554,432 elems per gate buffer

typedef __attribute__((ext_vector_type(16))) __bf16      v16bf;
typedef __attribute__((ext_vector_type(8)))  float       v8f;
typedef __attribute__((ext_vector_type(4)))  unsigned int v4u;

union Frag16 { v4u q[2]; v16bf v; };

// Load a 16x32 bf16 WMMA A/B fragment (row-major source, this lane's row):
// VGPR layout (ISA 7.12.2): lanes 0-15 K{0..7,16..23}, lanes 16-31 K{8..15,24..31}.
// Caller passes p = row_base + kbase where kbase = (lane>>4)*8.
__device__ __forceinline__ v16bf ldfrag(const __bf16* p) {
    Frag16 f;
    f.q[0] = *(const v4u*)(p);        // K kbase..kbase+7
    f.q[1] = *(const v4u*)(p + 16);   // K kbase+16..kbase+23
    return f.v;
}

__device__ __forceinline__ v8f wmma_bf16(v16bf a, v16bf b, v8f c) {
    return __builtin_amdgcn_wmma_f32_16x16x32_bf16(
        /*neg_a=*/false, a, /*neg_b=*/false, b,
        /*c_mod=*/(short)0, c, /*reuse_a=*/false, /*reuse_b=*/false);
}

struct Ptr6 { const float* p[6]; };

// ---------------------------------------------------------------------------
// Tensor Data Mover (gfx1250). 2D tile load, bf16 elements, LDS row padding
// 64B data + 16B pad (pad_interval code 3 = 16 DWORDs, pad_amount code 3 = 4
// DWORDs) to match the LDS_STRIDE=40 fragment layout.
// ---------------------------------------------------------------------------
#if __has_builtin(__builtin_amdgcn_tensor_load_to_lds)
#define USE_TDM 1
#if __has_include(<hip/amd_detail/amd_gfx1250_TDM.h>)
#define TDM_SIX_ARGS 1
#endif

typedef __attribute__((ext_vector_type(4))) unsigned int tdm_u4;
typedef __attribute__((ext_vector_type(8))) int          tdm_i8;
typedef __attribute__((ext_vector_type(4))) int          tdm_i4;

__device__ __forceinline__ void tdm_load_2d_bf16(unsigned lds_off,
                                                 const void* gptr,
                                                 unsigned rows,        // tile_dim1
                                                 unsigned rowlen,      // tile_dim0 (elems)
                                                 unsigned stride) {    // tensor_dim0_stride (elems)
    unsigned long long ga = (unsigned long long)(uintptr_t)gptr;
    tdm_u4 g0;
    g0[0] = 1u;                                   // count=1, user descriptor
    g0[1] = lds_off;                              // lds_addr (bytes)
    g0[2] = (unsigned)ga;                         // global_addr[31:0]
    g0[3] = (unsigned)((ga >> 32) & 0x01FFFFFFu)  // global_addr[56:32]
            | (2u << 30);                         // type = 2 ("image")
    tdm_i8 g1;
    g1[0] = (int)((1u << 16)                      // data_size = 1 -> 2 bytes
                | (1u << 20)                      // pad_enable
                | (3u << 22)                      // pad_interval: 16 DWORDs (64B)
                | (3u << 25));                    // pad_amount:   4 DWORDs (16B)
    g1[1] = (int)((rowlen & 0xFFFFu) << 16);      // tensor_dim0[15:0]
    g1[2] = (int)(((rowlen >> 16) & 0xFFFFu)      // tensor_dim0[31:16]
                | ((rows & 0xFFFFu) << 16));      // tensor_dim1[15:0]
    g1[3] = (int)(((rows >> 16) & 0xFFFFu)        // tensor_dim1[31:16]
                | ((rowlen & 0xFFFFu) << 16));    // tile_dim0
    g1[4] = (int)(rows & 0xFFFFu);                // tile_dim1 (tile_dim2 = 0)
    g1[5] = (int)stride;                          // tensor_dim0_stride[31:0]
    g1[6] = 0;                                    // stride0 hi / stride1 lo
    g1[7] = 0;                                    // stride1 hi
    tdm_i4 z4 = {0, 0, 0, 0};
#ifdef TDM_SIX_ARGS
    tdm_i8 z8 = {0, 0, 0, 0, 0, 0, 0, 0};
    __builtin_amdgcn_tensor_load_to_lds(g0, g1, z4, z4, z8, 0);
#else
    __builtin_amdgcn_tensor_load_to_lds(g0, g1, z4, z4, 0);
#endif
}
#endif // USE_TDM

// ---------------------------------------------------------------------------
// Conversion / init kernels
// ---------------------------------------------------------------------------
__global__ void conv_f32_to_bf16(const float* __restrict__ src,
                                 __bf16* __restrict__ dst, long n) {
    long i = (long)blockIdx.x * blockDim.x + threadIdx.x;
    if (i < n) dst[i] = (__bf16)src[i];
}

// Transpose + convert weights: src g is (K, H) row-major f32; dst is
// stacked (6, H, K) row-major bf16 ("N-major") so B fragments load contiguously.
__global__ void conv_w_transpose(Ptr6 src, __bf16* __restrict__ dst, int K) {
    long idx = (long)blockIdx.x * blockDim.x + threadIdx.x;
    long per = (long)K * GH;
    int  g = (int)(idx / per);
    long r = idx % per;
    int  n = (int)(r / K);
    int  k = (int)(r % K);
    dst[((size_t)g * GH + n) * K + k] = (__bf16)src.p[g][(size_t)k * GH + n];
}

__global__ void init_state(const float* __restrict__ hpf,
                           const float* __restrict__ hpb,
                           __bf16* __restrict__ hbf,
                           unsigned* __restrict__ bar) {
    int i = blockIdx.x * blockDim.x + threadIdx.x;
    if (i < 16) bar[i] = 0u;
    if (i < GB * GH) {
        hbf[i]           = (__bf16)hpf[i];
        hbf[GB * GH + i] = (__bf16)hpb[i];
    }
}

// ---------------------------------------------------------------------------
// Input-projection GEMM: (B*T, I) x (I, 6*H) with A bf16, Wt stacked (6H, I)
// N-major bf16. Output xproj[gate][t][b][h] f32 with bias added.
// Block = 256 thr (8 waves), tile 128x128, K staged 32 at a time.
// TDM double-buffered: wave 0 issues tensor_load_to_lds for the next tile,
// retires the previous pair via s_wait_tensorcnt, then the barrier releases
// the compute waves.
// ---------------------------------------------------------------------------
#define LDS_STRIDE 40   // 32 + pad, keeps 16B alignment for b128 LDS reads

__global__ __launch_bounds__(256)
void proj_gemm(const __bf16* __restrict__ A,    // (32768, 512)
               const __bf16* __restrict__ Wt,   // (6144, 512)
               float* __restrict__ xproj, Ptr6 bias) {
    __shared__ __bf16 sA[2][128 * LDS_STRIDE];
    __shared__ __bf16 sB[2][128 * LDS_STRIDE];

    const int tid  = threadIdx.x;
    const int wid  = tid >> 5;
    const int lane = tid & 31;
    const int m0   = blockIdx.x * 128;
    const int n0   = blockIdx.y * 128;
    const int r    = lane & 15;
    const int kofs = (lane >> 4) * 8;

    v8f acc[8] = {};

#ifdef USE_TDM
    const int NTILE = GI / 32;   // 16
    if (wid == 0) {
        tdm_load_2d_bf16((unsigned)(uintptr_t)&sA[0][0],
                         &A[(size_t)m0 * GI], 128, 32, GI);
        tdm_load_2d_bf16((unsigned)(uintptr_t)&sB[0][0],
                         &Wt[(size_t)n0 * GI], 128, 32, GI);
    }
    for (int it = 0; it < NTILE; ++it) {
        const int buf = it & 1;
        if (wid == 0) {
            if (it + 1 < NTILE) {
                // prefetch next 128x32 tiles into the other buffer
                tdm_load_2d_bf16((unsigned)(uintptr_t)&sA[buf ^ 1][0],
                                 &A[(size_t)m0 * GI + (it + 1) * 32], 128, 32, GI);
                tdm_load_2d_bf16((unsigned)(uintptr_t)&sB[buf ^ 1][0],
                                 &Wt[(size_t)n0 * GI + (it + 1) * 32], 128, 32, GI);
                __builtin_amdgcn_s_wait_tensorcnt(2);  // oldest pair (tile it) done
            } else {
                __builtin_amdgcn_s_wait_tensorcnt(0);
            }
        }
        __syncthreads();   // LDS tile `buf` visible to all waves

        v16bf af = ldfrag(&sA[buf][(wid * 16 + r) * LDS_STRIDE + kofs]);
#pragma unroll
        for (int nt = 0; nt < 8; ++nt) {
            v16bf bf = ldfrag(&sB[buf][(nt * 16 + r) * LDS_STRIDE + kofs]);
            acc[nt] = wmma_bf16(af, bf, acc[nt]);
        }
        __syncthreads();   // compute done before buf is re-filled next iteration
    }
#else
    for (int kb = 0; kb < GI; kb += 32) {
        for (int c = tid; c < 512; c += 256) {
            int row = c >> 2, seg = c & 3;
            *(v4u*)(&sA[0][row * LDS_STRIDE + seg * 8]) =
                *(const v4u*)(&A[(size_t)(m0 + row) * GI + kb + seg * 8]);
            *(v4u*)(&sB[0][row * LDS_STRIDE + seg * 8]) =
                *(const v4u*)(&Wt[(size_t)(n0 + row) * GI + kb + seg * 8]);
        }
        __syncthreads();
        v16bf af = ldfrag(&sA[0][(wid * 16 + r) * LDS_STRIDE + kofs]);
#pragma unroll
        for (int nt = 0; nt < 8; ++nt) {
            v16bf bf = ldfrag(&sB[0][(nt * 16 + r) * LDS_STRIDE + kofs]);
            acc[nt] = wmma_bf16(af, bf, acc[nt]);
        }
        __syncthreads();
    }
#endif

    // Epilogue: C tile layout — elem i -> row = i + 8*(lane>>4), col = lane&15.
    const int half = lane >> 4;
#pragma unroll
    for (int nt = 0; nt < 8; ++nt) {
        int n = n0 + nt * 16 + r;
        int g = n >> 10, h = n & (GH - 1);
        float bs = bias.p[g][h];
#pragma unroll
        for (int i = 0; i < 8; ++i) {
            int m = m0 + wid * 16 + i + half * 8;
            int b = m >> 9, t = m & (GT - 1);
            xproj[(size_t)g * TBH + (size_t)t * (GB * GH) + (size_t)b * GH + h] =
                acc[nt][i] + bs;
        }
    }
}

// ---------------------------------------------------------------------------
// Persistent recurrent scan. Grid = 2 teams (fwd/bwd) x 16 blocks; each block
// owns 64 columns of H. Device-scope sense-reversing barriers sequence steps.
// ---------------------------------------------------------------------------
__device__ __forceinline__ void team_barrier(unsigned* cnt, unsigned* gen,
                                             unsigned nb) {
    __syncthreads();
    if (threadIdx.x == 0) {
        __threadfence();
        unsigned g = __hip_atomic_load(gen, __ATOMIC_RELAXED,
                                       __HIP_MEMORY_SCOPE_AGENT);
        unsigned arrived = __hip_atomic_fetch_add(cnt, 1u, __ATOMIC_ACQ_REL,
                                                  __HIP_MEMORY_SCOPE_AGENT) + 1u;
        if (arrived == nb) {
            __hip_atomic_store(cnt, 0u, __ATOMIC_RELAXED,
                               __HIP_MEMORY_SCOPE_AGENT);
            __hip_atomic_fetch_add(gen, 1u, __ATOMIC_ACQ_REL,
                                   __HIP_MEMORY_SCOPE_AGENT);
        } else {
            while (__hip_atomic_load(gen, __ATOMIC_ACQUIRE,
                                     __HIP_MEMORY_SCOPE_AGENT) == g)
                __builtin_amdgcn_s_sleep(1);
        }
        __threadfence();
    }
    __syncthreads();
}

#define NBLK 16   // blocks per direction; 64 columns of H each

__global__ __launch_bounds__(256)
void gru_scan(const float* __restrict__ xproj,   // [6][T][B][H]
              const __bf16* __restrict__ WhT,    // [6][H][H]  (N-major)
              __bf16* __restrict__ hbf,          // [2][B][H]  bf16 h
              __bf16* __restrict__ rhbf,         // [2][B][H]  bf16 R*h
              const float* __restrict__ hpf,     // initial h fwd (B,H)
              const float* __restrict__ hpb,     // initial h bwd (B,H)
              float* __restrict__ out,           // (B,T,2H)
              unsigned* __restrict__ bar) {
    const int d   = blockIdx.x / NBLK;           // 0 = fwd, 1 = bwd
    const int blk = blockIdx.x % NBLK;
    const int n0  = blk * 64;                    // owned H columns [n0, n0+64)

    const int tid  = threadIdx.x;
    const int wid  = tid >> 5;
    const int lane = tid & 31;
    const int r    = lane & 15;
    const int half = lane >> 4;
    const int kofs = half * 8;

    unsigned* cnt = &bar[d * 8 + 0];
    unsigned* gen = &bar[d * 8 + 1];

    __bf16* hbf_d = hbf  + (size_t)d * GB * GH;
    __bf16* rh_d  = rhbf + (size_t)d * GB * GH;
    const float* xr = xproj + (size_t)(d * 3 + 0) * TBH;
    const float* xz = xproj + (size_t)(d * 3 + 1) * TBH;
    const float* xh = xproj + (size_t)(d * 3 + 2) * TBH;
    const float* h0 = d ? hpb : hpf;

    // Per-block f32 master copy of owned h slice, and Z gate scratch.
    __shared__ float hloc[GB * 64];
    __shared__ float zbuf[GB * 64];
    for (int i = tid; i < GB * 64; i += 256)
        hloc[i] = h0[(i >> 6) * GH + n0 + (i & 63)];
    // (no global h init needed here: init_state filled hbf)

    for (int s = 0; s < GT; ++s) {
        team_barrier(cnt, gen, NBLK);            // h (hbf) globally visible
        const int t = d ? (GT - 1 - s) : s;

        // ---- Phase 1: R,Z = sigmoid(h @ [W_hr | W_hz] + x{r,z}) -----------
        {
            const int gate = wid >> 2;           // 0 -> r, 1 -> z
            const int nt   = wid & 3;
            const int nc0  = n0 + nt * 16;
            const __bf16* Wrow =
                WhT + ((size_t)(d * 3 + gate) * GH + nc0 + r) * GH + kofs;
            v8f acc[4] = {};
            for (int kb = 0; kb < GH; kb += 32) {
                __builtin_prefetch(Wrow + kb + 256, 0, 1);
                v16bf bf = ldfrag(Wrow + kb);
#pragma unroll
                for (int mt = 0; mt < 4; ++mt) {
                    v16bf af = ldfrag(hbf_d + (size_t)(mt * 16 + r) * GH + kb + kofs);
                    acc[mt] = wmma_bf16(af, bf, acc[mt]);
                }
            }
            const float* xg = (gate == 0 ? xr : xz) + (size_t)t * (GB * GH);
            const int hcol = nc0 + r;
#pragma unroll
            for (int mt = 0; mt < 4; ++mt) {
#pragma unroll
                for (int i = 0; i < 8; ++i) {
                    int m = mt * 16 + i + half * 8;
                    float v = acc[mt][i] + xg[(size_t)m * GH + hcol];
                    v = 1.f / (1.f + __expf(-v));
                    if (gate == 0) {
                        float hp = hloc[m * 64 + (hcol - n0)];
                        rh_d[(size_t)m * GH + hcol] = (__bf16)(v * hp);
                    } else {
                        zbuf[m * 64 + (hcol - n0)] = v;
                    }
                }
            }
        }
        __threadfence();
        team_barrier(cnt, gen, NBLK);            // R*h globally visible

        // ---- Phase 2: h~ = tanh((R*h) @ W_hh + xh); h = Z*h~ + (1-Z)*h ----
        {
            const int nt  = wid & 3;
            const int mtb = (wid >> 2) * 2;
            const int nc0 = n0 + nt * 16;
            const __bf16* Wrow =
                WhT + ((size_t)(d * 3 + 2) * GH + nc0 + r) * GH + kofs;
            v8f acc[2] = {};
            for (int kb = 0; kb < GH; kb += 32) {
                __builtin_prefetch(Wrow + kb + 256, 0, 1);
                v16bf bf = ldfrag(Wrow + kb);
#pragma unroll
                for (int mm = 0; mm < 2; ++mm) {
                    v16bf af = ldfrag(rh_d + (size_t)((mtb + mm) * 16 + r) * GH + kb + kofs);
                    acc[mm] = wmma_bf16(af, bf, acc[mm]);
                }
            }
            const float* xh_t = xh + (size_t)t * (GB * GH);
            const int hcol = nc0 + r;
#pragma unroll
            for (int mm = 0; mm < 2; ++mm) {
#pragma unroll
                for (int i = 0; i < 8; ++i) {
                    int m = (mtb + mm) * 16 + i + half * 8;
                    float ht = tanhf(acc[mm][i] + xh_t[(size_t)m * GH + hcol]);
                    float z  = zbuf[m * 64 + (hcol - n0)];
                    float hp = hloc[m * 64 + (hcol - n0)];
                    float hn = z * ht + (1.f - z) * hp;
                    hloc[m * 64 + (hcol - n0)] = hn;
                    hbf_d[(size_t)m * GH + hcol] = (__bf16)hn;
                    out[(size_t)m * (GT * 2 * GH) + (size_t)t * (2 * GH) +
                        (size_t)d * GH + hcol] = hn;
                }
            }
        }
        __threadfence();
    }
}

// ---------------------------------------------------------------------------
// Host launcher
// ---------------------------------------------------------------------------
extern "C" void kernel_launch(void* const* d_in, const int* in_sizes, int n_in,
                              void* d_out, int out_size, void* d_ws, size_t ws_size,
                              hipStream_t stream) {
    const float* inputs = (const float*)d_in[0];
    const float* hpf    = (const float*)d_in[1];
    const float* hpb    = (const float*)d_in[2];

    // gate order g: 0 r_f, 1 z_f, 2 h_f, 3 r_b, 4 z_b, 5 h_b
    const int wh_idx[6] = {3, 6, 9, 12, 15, 18};
    const int wx_idx[6] = {4, 7, 10, 13, 16, 19};
    const int b_idx [6] = {5, 8, 11, 14, 17, 20};

    Ptr6 whp, wxp, bp;
    for (int g = 0; g < 6; ++g) {
        whp.p[g] = (const float*)d_in[wh_idx[g]];
        wxp.p[g] = (const float*)d_in[wx_idx[g]];
        bp.p[g]  = (const float*)d_in[b_idx[g]];
    }

    // Workspace layout (bytes)
    char* ws = (char*)d_ws;
    size_t off = 0;
    __bf16* Abf   = (__bf16*)(ws + off); off += (size_t)GB * GT * GI * 2;       // 32 MB
    __bf16* WxT   = (__bf16*)(ws + off); off += (size_t)6 * GH * GI * 2;        // 6 MB
    __bf16* WhT   = (__bf16*)(ws + off); off += (size_t)6 * GH * GH * 2;        // 12 MB
    float*  xproj = (float*) (ws + off); off += (size_t)6 * TBH * 4;            // 768 MB
    __bf16* hbf   = (__bf16*)(ws + off); off += (size_t)2 * GB * GH * 2;
    __bf16* rhbf  = (__bf16*)(ws + off); off += (size_t)2 * GB * GH * 2;
    unsigned* bar = (unsigned*)(ws + off); off += 256;

    // 1) inputs f32 -> bf16
    {
        long n = (long)GB * GT * GI;
        conv_f32_to_bf16<<<(int)(n / 256), 256, 0, stream>>>(inputs, Abf, n);
    }
    // 2) transpose+convert Wx (I,H) -> (6,H,I) and Wh (H,H) -> (6,H,H)
    conv_w_transpose<<<(6 * GI * GH) / 256, 256, 0, stream>>>(wxp, WxT, GI);
    conv_w_transpose<<<(6 * GH * GH) / 256, 256, 0, stream>>>(whp, WhT, GH);
    // 3) init h bf16 + barrier state
    init_state<<<256, 256, 0, stream>>>(hpf, hpb, hbf, bar);
    // 4) fused input-projection GEMM (M=32768, K=512, N=6*1024) + bias
    proj_gemm<<<dim3((GB * GT) / 128, (6 * GH) / 128), 256, 0, stream>>>(
        Abf, WxT, xproj, bp);
    // 5) persistent bidirectional scan
    gru_scan<<<2 * NBLK, 256, 0, stream>>>(xproj, WhT, hbf, rhbf, hpf, hpb,
                                           (float*)d_out, bar);
}
